// Model_74371653698308
// MI455X (gfx1250) — matrix operand order repaired and checked by reading, not compiled, not run on hardware
//
#include <hip/hip_runtime.h>
#include <hip/hip_bf16.h>

// Persistent-RNN LSTM for MI455X (gfx1250, wave32, WMMA).
// One workgroup per 16-row batch tile runs the whole T=1024 recurrence.
// Weights live in VGPRs as pre-swizzled f16 WMMA B-fragments.

typedef __attribute__((ext_vector_type(16))) _Float16 v16h;
typedef __attribute__((ext_vector_type(8)))  _Float16 v8h;
typedef __attribute__((ext_vector_type(8)))  float    v8f;

#define T_SEQ  1024
#define EMB    128
#define HID    128
#define G4H    512     // 4*HID
#define BTILE  16      // batch rows per workgroup (== WMMA M)
#define ZPAD   20      // fp32 stride per z column: 16 rows + pad (80B, keeps 16B align, breaks bank conflicts)

__device__ __forceinline__ float fast_sigmoid(float z) {
    return __builtin_amdgcn_rcpf(1.0f + __expf(-z));
}
__device__ __forceinline__ float fast_tanh(float z) {
    return 1.0f - 2.0f * __builtin_amdgcn_rcpf(1.0f + __expf(2.0f * z));
}

__launch_bounds__(256, 1)
__global__ void lstm_persistent_kernel(const int*   __restrict__ texts,
                                       const float* __restrict__ emb,
                                       const float* __restrict__ Wk,
                                       const float* __restrict__ Wr,
                                       const float* __restrict__ bias,
                                       const float* __restrict__ Wd,
                                       const float* __restrict__ bd,
                                       float*       __restrict__ out)
{
    __shared__ _Float16 sh_x[BTILE * EMB];   // x_t tile, f16, row-major (A-fragment friendly)
    __shared__ _Float16 sh_h[BTILE * HID];   // h tile,  f16, row-major
    __shared__ float    sh_z[G4H * ZPAD];    // z staged transposed: z[col][row]

    const int tid    = threadIdx.x;
    const int lane   = tid & 31;
    const int wave   = tid >> 5;             // 0..7; wave owns gate columns [wave*64, wave*64+64)
    const int batch0 = blockIdx.x * BTILE;

    // ---------------- one-time: weights -> f16 WMMA B-fragments in VGPRs -----------
    // B 32x16 (KxN) fragment: lane L holds column c = ntile*16 + (L&15),
    // K = ktile*32 + (L<16 ? 0..15 : 16..31), contiguous in K within the v16h.
    const int bcol  = lane & 15;
    const int bksel = (lane >> 4) << 4;      // 0 or 16
    v16h wkf[4][4], wrf[4][4];
#pragma unroll
    for (int n = 0; n < 4; ++n) {
        const int c = (wave * 4 + n) * 16 + bcol;
#pragma unroll
        for (int k = 0; k < 4; ++k) {
            const int kb = k * 32 + bksel;
            v16h wk, wr;
#pragma unroll
            for (int e = 0; e < 16; ++e) {
                wk[e] = (_Float16)Wk[(kb + e) * G4H + c];
                wr[e] = (_Float16)Wr[(kb + e) * G4H + c];
            }
            wkf[n][k] = wk;
            wrf[n][k] = wr;
        }
    }
    // bias per n-tile (C/D layout: N = lane&15, so bias depends only on column)
    float bv[4];
#pragma unroll
    for (int n = 0; n < 4; ++n) bv[n] = bias[(wave * 4 + n) * 16 + bcol];

    // A-fragment addressing (16x32 f16 A layout):
    // lane L: row = L&15; elems 0..15 = K kb..kb+15 (contiguous, matches B),
    // with kb = ktile*32 + (L<16 ? 0 : 16).
    const int arow  = lane & 15;
    const int aksel = (lane >> 4) << 4;      // 0 or 16

    // gate-phase mapping: thread -> (row grow, hidden units gj0..gj0+7); c stays in registers
    const int grow = tid >> 4;               // 0..15
    const int gj0  = (tid & 15) * 8;
    float creg[8];
#pragma unroll
    for (int e = 0; e < 8; ++e) creg[e] = 0.0f;

    // h0 = 0
#pragma unroll
    for (int e = 0; e < 8; ++e) sh_h[tid * 8 + e] = (_Float16)0.0f;
    __syncthreads();

    for (int t = 0; t < T_SEQ; ++t) {
        // ---- stage x_t = emb[texts[:, t]] into LDS as f16 ----
        {
            const int row = tid >> 4;
            const int c8  = (tid & 15) * 8;
            const int idx = texts[(batch0 + row) * T_SEQ + t];
            const float4 f0 = *(const float4*)(emb + (size_t)idx * EMB + c8);
            const float4 f1 = *(const float4*)(emb + (size_t)idx * EMB + c8 + 4);
            v8h hx;
            hx[0] = (_Float16)f0.x; hx[1] = (_Float16)f0.y;
            hx[2] = (_Float16)f0.z; hx[3] = (_Float16)f0.w;
            hx[4] = (_Float16)f1.x; hx[5] = (_Float16)f1.y;
            hx[6] = (_Float16)f1.z; hx[7] = (_Float16)f1.w;
            *(v8h*)(sh_x + row * EMB + c8) = hx;
            // prefetch next timestep's embedding rows (global_prefetch_b8)
            if ((tid & 15) == 0 && (t + 1) < T_SEQ) {
                const int idx2 = texts[(batch0 + row) * T_SEQ + t + 1];
                __builtin_prefetch(emb + (size_t)idx2 * EMB, 0, 1);
            }
        }
        __syncthreads();

        // ---- z = x_t @ Wk + h @ Wr + b  (per wave: 4 n-tiles, K=128 in 4 WMMAs each) ----
        v16h ax[4], ah[4];
#pragma unroll
        for (int k = 0; k < 4; ++k) {
            const int kb = k * 32 + aksel;
            v8h lo = *(const v8h*)(sh_x + arow * EMB + kb);
            v8h hi = *(const v8h*)(sh_x + arow * EMB + kb + 8);
#pragma unroll
            for (int e = 0; e < 8; ++e) { ax[k][e] = lo[e]; ax[k][e + 8] = hi[e]; }
            lo = *(const v8h*)(sh_h + arow * HID + kb);
            hi = *(const v8h*)(sh_h + arow * HID + kb + 8);
#pragma unroll
            for (int e = 0; e < 8; ++e) { ah[k][e] = lo[e]; ah[k][e + 8] = hi[e]; }
        }
#pragma unroll
        for (int n = 0; n < 4; ++n) {
            v8f acc;
#pragma unroll
            for (int e = 0; e < 8; ++e) acc[e] = bv[n];
#pragma unroll
            for (int k = 0; k < 4; ++k)
                acc = __builtin_amdgcn_wmma_f32_16x16x32_f16(
                        false, ax[k], false, wkf[n][k], (short)0, acc, false, false);
#pragma unroll
            for (int k = 0; k < 4; ++k)
                acc = __builtin_amdgcn_wmma_f32_16x16x32_f16(
                        false, ah[k], false, wrf[n][k], (short)0, acc, false, false);
            // C/D layout: N = lane&15, M = (lane<16 ? 0 : 8) + vgprIdx. Store transposed z[col][row].
            const int nabs = (wave * 4 + n) * 16 + bcol;
            const int m0   = (lane >> 4) * 8;
            float4 z0 = { acc[0], acc[1], acc[2], acc[3] };
            float4 z1 = { acc[4], acc[5], acc[6], acc[7] };
            *(float4*)(sh_z + nabs * ZPAD + m0)     = z0;
            *(float4*)(sh_z + nabs * ZPAD + m0 + 4) = z1;
        }
        __syncthreads();

        // ---- gates: i,f,g,o -> c,h update (Keras gate order i,f,c,o) ----
        {
            v8h hrow;
#pragma unroll
            for (int e = 0; e < 8; ++e) {
                const int j  = gj0 + e;
                const float zi = sh_z[(j          ) * ZPAD + grow];
                const float zf = sh_z[(j + HID    ) * ZPAD + grow];
                const float zg = sh_z[(j + 2 * HID) * ZPAD + grow];
                const float zo = sh_z[(j + 3 * HID) * ZPAD + grow];
                const float ig = fast_sigmoid(zi);
                const float fg = fast_sigmoid(zf);
                const float gg = fast_tanh(zg);
                const float og = fast_sigmoid(zo);
                creg[e] = fg * creg[e] + ig * gg;
                hrow[e] = (_Float16)(og * fast_tanh(creg[e]));
            }
            *(v8h*)(sh_h + grow * HID + gj0) = hrow;   // 16B contiguous
        }
        __syncthreads();
    }

    // logits = h @ Wd + bd is a [B,1] vector; softmax over a size-1 axis == 1.0 exactly,
    // which is precisely what the reference returns.
    if (tid < BTILE) {
        out[batch0 + tid] = 1.0f;
    }
    (void)Wd; (void)bd;
}

extern "C" void kernel_launch(void* const* d_in, const int* in_sizes, int n_in,
                              void* d_out, int out_size, void* d_ws, size_t ws_size,
                              hipStream_t stream) {
    const int*   texts = (const int*)  d_in[0];
    const float* emb   = (const float*)d_in[1];
    const float* Wk    = (const float*)d_in[2];
    const float* Wr    = (const float*)d_in[3];
    const float* b     = (const float*)d_in[4];
    const float* Wd    = (const float*)d_in[5];
    const float* bd    = (const float*)d_in[6];
    float* out = (float*)d_out;

    const int B = in_sizes[0] / T_SEQ;       // 256
    dim3 grid(B / BTILE);                    // 16 workgroups, one per batch tile
    dim3 block(256);                         // 8 wave32 waves
    lstm_persistent_kernel<<<grid, block, 0, stream>>>(texts, emb, Wk, Wr, b, Wd, bd, out);
}
